// FakedditGNN_4544075399710
// MI455X (gfx1250) — matrix operand order repaired and checked by reading, not compiled, not run on hardware
//
#include <hip/hip_runtime.h>
#include <hip/hip_bf16.h>

// ---------------------------------------------------------------------------
// Fakeddit GNN: 3x GCNConv(+ReLU) + linear classifier + log_softmax
// N=100000 nodes, E=1.6M edges, F_IN=256, HID=128, N_CLS=2
// ---------------------------------------------------------------------------

typedef __attribute__((ext_vector_type(16))) __bf16 bf16x16;
typedef __attribute__((ext_vector_type(8)))  float  f32x8;
typedef __attribute__((address_space(3))) unsigned short lds_u16;

#define HID 128

__device__ __forceinline__ unsigned short f2bf(float f) {
  unsigned int u = __float_as_uint(f);
  unsigned int r = u + 0x7FFFu + ((u >> 16) & 1u);  // round-to-nearest-even
  return (unsigned short)(r >> 16);
}
__device__ __forceinline__ unsigned int pack2(float lo, float hi) {
  return (unsigned int)f2bf(lo) | ((unsigned int)f2bf(hi) << 16);
}

// ---------------------------------------------------------------------------
// One-shot per layer: WTg[col*K + k] = bf16(W[k*128 + col])   (transposed)
// ---------------------------------------------------------------------------
__global__ void prep_wt(const float* __restrict__ W,
                        unsigned short* __restrict__ WTg, int K) {
  int i = blockIdx.x * blockDim.x + threadIdx.x;
  if (i < K * HID) {
    int k   = i >> 7;       // i / 128  (coalesced read of W)
    int col = i & 127;
    WTg[col * K + k] = f2bf(W[i]);
  }
}

// ---------------------------------------------------------------------------
// GEMM: XW[N,128] = X[N,K] @ W[K,128], bf16 WMMA with f32 accumulate.
// Block = 256 threads = 8 waves; wave w computes rows [blk*128 + 16w, +16),
// all 128 output columns (8 accumulators of 16x16).
// Weights staged to LDS with gfx1250 async global->LDS copies.
// ---------------------------------------------------------------------------
__global__ __launch_bounds__(256) void gemm_xw_wmma(
    const float* __restrict__ X, const unsigned short* __restrict__ WTg,
    float* __restrict__ XW, int n, int K) {
  __shared__ __align__(16) unsigned short WT[128 * 256];  // <= 64KB used

  const int tid = threadIdx.x;

  // ---- async-copy bf16 W^T into LDS (ASYNCcnt-tracked DMA path) ----
  {
    const unsigned lds_base = (unsigned)(unsigned long long)(lds_u16*)WT;
    const int chunks = (HID * K * 2) / 16;  // 16B chunks
    for (int i = tid; i < chunks; i += 256) {
      unsigned laddr = lds_base + (unsigned)i * 16u;
      const char* g = (const char*)WTg + (size_t)i * 16u;
      asm volatile("global_load_async_to_lds_b128 %0, %1, off"
                   :: "v"(laddr), "v"(g) : "memory");
    }
    asm volatile("s_wait_asynccnt 0" ::: "memory");
  }
  __syncthreads();

  const int wave = tid >> 5;
  const int lane = tid & 31;
  const int h    = lane >> 4;   // lane half (K-group selector)
  const int mn   = lane & 15;   // row (A) / column-in-tile (B, C/D)
  const int row0 = blockIdx.x * 128 + wave * 16;

  int arow_i = row0 + mn;
  if (arow_i > n - 1) arow_i = n - 1;           // clamp tail loads
  const long long arow = (long long)arow_i * K;

  f32x8 acc[8];
  for (int t = 0; t < 8; ++t) acc[t] = (f32x8){0.f,0.f,0.f,0.f,0.f,0.f,0.f,0.f};

  for (int kb = 0; kb < K; kb += 32) {
    // ---- A fragment: 16-bit A 16x32 layout.
    // lane half h covers k-offsets {h*16 .. h*16+15} (contiguous, matches B)
    union { unsigned int u[8]; bf16x16 v; } a;
    const float4* p0 = reinterpret_cast<const float4*>(X + arow + kb + h * 16);
    const float4* p1 = reinterpret_cast<const float4*>(X + arow + kb + h * 16 + 8);
    float4 r0 = p0[0], r1 = p0[1];
    float4 r2 = p1[0], r3 = p1[1];
    a.u[0] = pack2(r0.x, r0.y); a.u[1] = pack2(r0.z, r0.w);
    a.u[2] = pack2(r1.x, r1.y); a.u[3] = pack2(r1.z, r1.w);
    a.u[4] = pack2(r2.x, r2.y); a.u[5] = pack2(r2.z, r2.w);
    a.u[6] = pack2(r3.x, r3.y); a.u[7] = pack2(r3.z, r3.w);

    for (int nt = 0; nt < 8; ++nt) {
      // ---- B fragment: 16-bit B 32x16 layout. Lane half h covers
      // k = kb + h*16 .. +15 for column nt*16+mn -> 32 contiguous LDS bytes.
      union { uint4 q[2]; bf16x16 v; } b;
      const uint4* bp = reinterpret_cast<const uint4*>(
          &WT[(nt * 16 + mn) * K + kb + h * 16]);
      b.q[0] = bp[0];
      b.q[1] = bp[1];
      acc[nt] = __builtin_amdgcn_wmma_f32_16x16x32_bf16(
          false, a.v, false, b.v, (short)0, acc[nt], false, false);
    }
  }

  // ---- C/D layout: VGPR r -> M = r (lanes 0-15) or 8+r (lanes 16-31); N = mn
  if (row0 + 16 <= n) {
    // full tile: unconditional stores
    float* base = XW + (long long)(row0 + h * 8) * HID + mn;
    for (int nt = 0; nt < 8; ++nt) {
      for (int r = 0; r < 8; ++r) {
        base[(long long)r * HID + nt * 16] = acc[nt][r];
      }
    }
  } else {
    for (int nt = 0; nt < 8; ++nt) {
      for (int r = 0; r < 8; ++r) {
        int row = row0 + h * 8 + r;
        if (row < n) XW[(long long)row * HID + nt * 16 + mn] = acc[nt][r];
      }
    }
  }
}

// ---------------------------------------------------------------------------
// Degree / normalization
// ---------------------------------------------------------------------------
__global__ void init_deg(float* __restrict__ deg, int n) {
  int i = blockIdx.x * blockDim.x + threadIdx.x;
  if (i < n) deg[i] = 1.0f;  // self-loop
}
__global__ void deg_count(const long long* __restrict__ dst,
                          float* __restrict__ deg, int E) {
  int e = blockIdx.x * blockDim.x + threadIdx.x;
  if (e < E) unsafeAtomicAdd(&deg[(int)dst[e]], 1.0f);
}
__global__ void deg_finalize(const float* __restrict__ deg,
                             float* __restrict__ dinv,
                             float* __restrict__ invdeg, int n) {
  int i = blockIdx.x * blockDim.x + threadIdx.x;
  if (i < n) {
    float d = deg[i];
    dinv[i]   = rsqrtf(d);
    invdeg[i] = 1.0f / d;
  }
}

// AGG[i] = XW[i] * invdeg[node]  (self-loop contribution, also zero-inits AGG)
__global__ void init_agg(const float* __restrict__ XW,
                         const float* __restrict__ invdeg,
                         float* __restrict__ AGG, long long total) {
  long long i = (long long)blockIdx.x * blockDim.x + threadIdx.x;
  if (i < total) AGG[i] = XW[i] * invdeg[i >> 7];
}

// Per (edge, 4-col chunk): AGG[dst] += XW[src] * dinv[src]*dinv[dst]
// One wave per edge (32 lanes x 4 cols = 128): coalesced float4 row gathers.
__global__ void scatter_edges(const long long* __restrict__ src,
                              const long long* __restrict__ dst,
                              const float* __restrict__ dinv,
                              const float* __restrict__ XW,
                              float* __restrict__ AGG, int E) {
  int gid = blockIdx.x * blockDim.x + threadIdx.x;
  int e = gid >> 5;
  if (e >= E) return;
  int c = (gid & 31) << 2;
  int s = (int)src[e];
  int d = (int)dst[e];
  float w = dinv[s] * dinv[d];
  float4 v = *reinterpret_cast<const float4*>(XW + (long long)s * HID + c);
  float* out = AGG + (long long)d * HID + c;
  unsafeAtomicAdd(out + 0, v.x * w);
  unsafeAtomicAdd(out + 1, v.y * w);
  unsafeAtomicAdd(out + 2, v.z * w);
  unsafeAtomicAdd(out + 3, v.w * w);
}

// H[i] = relu(AGG[i] + b[col])
__global__ void bias_relu(const float* __restrict__ AGG,
                          const float* __restrict__ b,
                          float* __restrict__ H, long long total) {
  long long i = (long long)blockIdx.x * blockDim.x + threadIdx.x;
  if (i < total) H[i] = fmaxf(AGG[i] + b[i & (HID - 1)], 0.0f);
}

// logits = H @ Wc + bc ; out = log_softmax(logits)
__global__ void classify_logsoftmax(const float* __restrict__ H,
                                    const float* __restrict__ Wc,
                                    const float* __restrict__ bc,
                                    float* __restrict__ out, int n) {
  int i = blockIdx.x * blockDim.x + threadIdx.x;
  if (i >= n) return;
  const float* h = H + (long long)i * HID;
  float s0 = bc[0], s1 = bc[1];
  for (int k = 0; k < HID; ++k) {
    float v = h[k];
    s0 += v * Wc[2 * k];
    s1 += v * Wc[2 * k + 1];
  }
  float m = fmaxf(s0, s1);
  float l = logf(expf(s0 - m) + expf(s1 - m));
  out[2 * i + 0] = s0 - m - l;
  out[2 * i + 1] = s1 - m - l;
}

// ---------------------------------------------------------------------------
extern "C" void kernel_launch(void* const* d_in, const int* in_sizes, int n_in,
                              void* d_out, int out_size, void* d_ws, size_t ws_size,
                              hipStream_t stream) {
  const float*     x   = (const float*)d_in[0];
  const long long* ei  = (const long long*)d_in[1];  // int64 [2, E]
  const float*     W1  = (const float*)d_in[2];
  const float*     b1  = (const float*)d_in[3];
  const float*     W2  = (const float*)d_in[4];
  const float*     b2  = (const float*)d_in[5];
  const float*     W3  = (const float*)d_in[6];
  const float*     b3  = (const float*)d_in[7];
  const float*     Wc  = (const float*)d_in[8];
  const float*     bc  = (const float*)d_in[9];
  float*           out = (float*)d_out;

  const int K1 = in_sizes[2] / HID;       // 256
  const int n  = in_sizes[0] / K1;        // 100000
  const int E  = in_sizes[1] / 2;         // 1600000
  const long long NH = (long long)n * HID;

  const long long* src = ei;
  const long long* dst = ei + E;

  // workspace carve-up
  float*          XW     = (float*)d_ws;
  float*          AGG    = XW  + NH;
  float*          H      = AGG + NH;
  float*          deg    = H   + NH;
  float*          dinv   = deg + n;
  float*          invdeg = dinv + n;
  unsigned short* WTg    = (unsigned short*)(invdeg + n);  // 128*256 bf16

  const int TB = 256;
  const int gN   = (n + TB - 1) / TB;
  const int gE   = (E + TB - 1) / TB;
  const int gNH  = (int)((NH + TB - 1) / TB);
  const int gSC  = (int)(((long long)E * 32 + TB - 1) / TB);
  const int gMM  = (n + 127) / 128;
  const int gW1  = (K1 * HID + TB - 1) / TB;
  const int gW2  = (HID * HID + TB - 1) / TB;

  // degree / normalization (shared by all layers)
  init_deg<<<gN, TB, 0, stream>>>(deg, n);
  deg_count<<<gE, TB, 0, stream>>>(dst, deg, E);
  deg_finalize<<<gN, TB, 0, stream>>>(deg, dinv, invdeg, n);

  // ---- layer 1 ----
  prep_wt<<<gW1, TB, 0, stream>>>(W1, WTg, K1);
  gemm_xw_wmma<<<gMM, TB, 0, stream>>>(x, WTg, XW, n, K1);
  init_agg<<<gNH, TB, 0, stream>>>(XW, invdeg, AGG, NH);
  scatter_edges<<<gSC, TB, 0, stream>>>(src, dst, dinv, XW, AGG, E);
  bias_relu<<<gNH, TB, 0, stream>>>(AGG, b1, H, NH);

  // ---- layer 2 ----
  prep_wt<<<gW2, TB, 0, stream>>>(W2, WTg, HID);
  gemm_xw_wmma<<<gMM, TB, 0, stream>>>(H, WTg, XW, n, HID);
  init_agg<<<gNH, TB, 0, stream>>>(XW, invdeg, AGG, NH);
  scatter_edges<<<gSC, TB, 0, stream>>>(src, dst, dinv, XW, AGG, E);
  bias_relu<<<gNH, TB, 0, stream>>>(AGG, b2, H, NH);

  // ---- layer 3 ----
  prep_wt<<<gW2, TB, 0, stream>>>(W3, WTg, HID);
  gemm_xw_wmma<<<gMM, TB, 0, stream>>>(H, WTg, XW, n, HID);
  init_agg<<<gNH, TB, 0, stream>>>(XW, invdeg, AGG, NH);
  scatter_edges<<<gSC, TB, 0, stream>>>(src, dst, dinv, XW, AGG, E);
  bias_relu<<<gNH, TB, 0, stream>>>(AGG, b3, H, NH);

  // ---- classifier + log_softmax ----
  classify_logsoftmax<<<gN, TB, 0, stream>>>(H, Wc, bc, out, n);
}
